// SimpleQSANIModel_21878563406028
// MI455X (gfx1250) — compile-verified, matrix operand order checked by reading, not run on hardware
//
#include <hip/hip_runtime.h>

// ---------------------------------------------------------------------------
// Problem constants (from reference)
// ---------------------------------------------------------------------------
#define NMOL        2048
#define NATOM       64
#define AEV_LEN     384
#define FEAT_PAD    416           // 386 padded up to a multiple of 32 for WMMA K
#define NATOMS      (NMOL * NATOM)
#define A0_CONST    0.529177249f

typedef __attribute__((ext_vector_type(16))) _Float16     v16h;
typedef __attribute__((ext_vector_type(8)))  float        v8f;
typedef __attribute__((ext_vector_type(4)))  unsigned int u32x4;
typedef __attribute__((ext_vector_type(4)))  int          i32x4;
typedef __attribute__((ext_vector_type(8)))  int          i32x8;

union FragH { v16h v; uint4 q[2]; };

// ---------------------------------------------------------------------------
// WMMA fragment loads (wave32, 16x16x32 f16 -> f32)
// ---------------------------------------------------------------------------
// A-matrix (16xK f16) fragment from LDS, row-major with row stride = din halves.
// Per ISA 7.12.2: lanes 0-15 hold K {0..7,16..23}, lanes 16-31 hold K {8..15,24..31}
__device__ __forceinline__ v16h load_a_lds(const _Float16* X, int din, int ks, int lane) {
    const int hf = lane >> 4;
    const int m  = lane & 15;
    const _Float16* p = X + m * din + ks * 32 + hf * 8;   // halves
    FragH f;
    f.q[0] = *(const uint4*)(p);
    f.q[1] = *(const uint4*)(p + 16);
    return f.v;
}

// B-matrix fragment from LDS-staged swizzled weights: each (kstep, ntile) block is
// 512 halves, lane-major (lane*16 + elem) -> two ds_load_b128 per lane.
__device__ __forceinline__ v16h load_b_lds(const _Float16* Wlds, int blockLin, int lane) {
    const uint4* p = (const uint4*)(Wlds + (size_t)blockLin * 512 + lane * 16);
    FragH f;
    f.q[0] = p[0];
    f.q[1] = p[1];
    return f.v;
}

__device__ __forceinline__ float celu01(float x) {
    return x > 0.f ? x : 0.1f * (__expf(x * 10.f) - 1.f);
}

// ---------------------------------------------------------------------------
// Weight staging paths
// ---------------------------------------------------------------------------
// (a) TDM: one descriptor moves the whole blob (2-D tile: 512 dwords x rows),
//     issued by a single wave, tracked by TENSORcnt.
__device__ __forceinline__ void tdm_load_blob(const void* gsrc, void* ldsDst,
                                              unsigned dwords) {
    const unsigned rows = dwords >> 9;                  // blob is k*512 dwords
    const unsigned long long ga = (unsigned long long)(uintptr_t)gsrc;
    const unsigned lds = (unsigned)(uintptr_t)ldsDst;
    u32x4 g0;
    g0[0] = 1u;                                         // count=1 (valid user D#)
    g0[1] = lds;                                        // lds_addr
    g0[2] = (unsigned)(ga & 0xffffffffu);               // global_addr[31:0]
    g0[3] = (unsigned)((ga >> 32) & 0x01ffffffu)        // global_addr[56:32]
          | (2u << 30);                                 // type = 2 ("image")
    i32x8 g1;
    g1[0] = (int)(2u << 16);                            // data_size = 4 bytes
    g1[1] = (int)(512u << 16);                          // tensor_dim0 = 512 (lo16)
    g1[2] = (int)(rows << 16);                          // dim0 hi=0 | tensor_dim1 lo
    g1[3] = (int)(512u << 16);                          // dim1 hi=0 | tile_dim0 = 512
    g1[4] = (int)(rows & 0xffffu);                      // tile_dim1 = rows, tile_dim2=0
    g1[5] = (int)512u;                                  // tensor_dim0_stride = 512
    g1[6] = 0;
    g1[7] = 0;
    const i32x4 z4 = {0, 0, 0, 0};
#if defined(__clang_major__) && (__clang_major__ >= 23)
    const i32x8 z8 = {0, 0, 0, 0, 0, 0, 0, 0};
    __builtin_amdgcn_tensor_load_to_lds(g0, g1, z4, z4, z8, 0);
#else
    __builtin_amdgcn_tensor_load_to_lds(g0, g1, z4, z4, 0);
#endif
}

// (b) Async global->LDS b128 loop (ASYNCcnt) for the smaller layers.
__device__ __forceinline__ void stage_weights_async(const _Float16* __restrict__ src,
                                                    _Float16* dst, int halves,
                                                    int tid, int nthreads) {
    const int chunks = halves >> 3;                    // 16-byte chunks
    for (int c = tid; c < chunks; c += nthreads) {
        unsigned      ldsAddr = (unsigned)(uintptr_t)(dst + (size_t)c * 8);
        const void*   gAddr   = (const void*)(src + (size_t)c * 8);
        asm volatile("global_load_async_to_lds_b128 %0, %1, off"
                     :: "v"(ldsAddr), "v"(gAddr) : "memory");
    }
    asm volatile("s_wait_asynccnt 0" ::: "memory");    // per-wave drain of ASYNCcnt
}

// ---------------------------------------------------------------------------
// One dense layer via WMMA: H(16 x DOUT) = act(X(16 x DIN) @ W + b).
// All A fragments cached in registers (KT x 8 VGPRs); N tiles processed in
// pairs with two independent accumulator chains (hides WMMA->WMMA hazards).
// ---------------------------------------------------------------------------
template<int KT, int NT, bool ACT>
__device__ __forceinline__ void wmma_layer_t(const _Float16* Xlds,
                                             const _Float16* Wlds,
                                             const float* __restrict__ bias,
                                             _Float16* Hlds, int lane) {
    constexpr int DIN  = KT * 32;
    constexpr int DOUT = NT * 16;
    const int n  = lane & 15;
    const int hf = lane >> 4;
    v16h a[KT];
#pragma unroll
    for (int ks = 0; ks < KT; ++ks) a[ks] = load_a_lds(Xlds, DIN, ks, lane);
#pragma unroll 1
    for (int nt = 0; nt < NT; nt += 2) {
        v8f acc0 = {0.f, 0.f, 0.f, 0.f, 0.f, 0.f, 0.f, 0.f};
        v8f acc1 = {0.f, 0.f, 0.f, 0.f, 0.f, 0.f, 0.f, 0.f};
#pragma unroll
        for (int ks = 0; ks < KT; ++ks) {
            v16h b0 = load_b_lds(Wlds, ks * NT + nt,     lane);
            v16h b1 = load_b_lds(Wlds, ks * NT + nt + 1, lane);
            acc0 = __builtin_amdgcn_wmma_f32_16x16x32_f16(
                false, a[ks], false, b0, (short)0, acc0, false, false);
            acc1 = __builtin_amdgcn_wmma_f32_16x16x32_f16(
                false, a[ks], false, b1, (short)0, acc1, false, false);
        }
        const float bv0 = bias[nt * 16 + n];
        const float bv1 = bias[nt * 16 + 16 + n];
#pragma unroll
        for (int i = 0; i < 8; ++i) {
            float y0 = acc0[i] + bv0;
            float y1 = acc1[i] + bv1;
            if (ACT) { y0 = celu01(y0); y1 = celu01(y1); }
            const int m = i + 8 * hf;
            Hlds[m * DOUT + nt * 16 + n]      = (_Float16)y0;
            Hlds[m * DOUT + nt * 16 + 16 + n] = (_Float16)y1;
        }
    }
    asm volatile("s_wait_dscnt 0" ::: "memory");  // H stores visible to next layer
}

// ---------------------------------------------------------------------------
// Kernel 0: swizzle f32 weights into WMMA B-fragment order (f16), K zero-padded.
// dst[((ks*nT + nt)*32 + lane)*16 + e] = W[r, nt*16 + n],
//   r = ks*32 + 2v + (v>=4 ? 8 : 0) + 8*half + lo,  v=e>>1, lo=e&1
// ---------------------------------------------------------------------------
__global__ void swizzle_weights_kernel(const float* __restrict__ W,
                                       _Float16* __restrict__ dst,
                                       int din, int dout, int kSteps) {
    const int nT = dout >> 4;
    const int total = kSteps * nT * 512;
    int gid = blockIdx.x * blockDim.x + threadIdx.x;
    if (gid >= total) return;
    const int e    = gid & 15;
    const int lane = (gid >> 4) & 31;
    const int nt   = (gid >> 9) % nT;
    const int ks   = gid / (512 * nT);
    const int n    = lane & 15;
    const int hf   = lane >> 4;
    const int v    = e >> 1;
    const int lo   = e & 1;
    const int r    = ks * 32 + 2 * v + ((v >= 4) ? 8 : 0) + 8 * hf + lo;
    const float val = (r < din) ? W[r * dout + (nt * 16 + n)] : 0.f;
    dst[gid] = (_Float16)val;
}

// ---------------------------------------------------------------------------
// Kernel 1: q-net MLP (384 -> 160 -> 128 -> 96 -> 1) for all atoms.
// 8 waves/WG = 128 atoms; layer-1 weights staged by TDM (overlapped with the
// activation load/convert), layers 2/3 by async global->LDS.
// LDS: 120 KB weight stage + 8 x 17 KB activation buffers = 256 KB.
// ---------------------------------------------------------------------------
#define QW_STAGE_HALVES   61440                       // max layer: 12*10*512
#define QX_PER_WAVE       (16 * AEV_LEN + 16 * 160)   // 8704 halves

__global__ __launch_bounds__(256) void q_mlp_kernel(
    const float* __restrict__ aev, const int* __restrict__ species,
    const _Float16* __restrict__ w0s, const float* __restrict__ b0,
    const _Float16* __restrict__ w1s, const float* __restrict__ b1,
    const _Float16* __restrict__ w2s, const float* __restrict__ b2,
    const float* __restrict__ w3, const float* __restrict__ b3,
    float* __restrict__ qAll) {
    extern __shared__ char smem[];
    _Float16* wLds = (_Float16*)smem;                          // shared weight stage
    const int tid  = threadIdx.x;
    const int wave = tid >> 5;
    const int lane = tid & 31;
    _Float16* bufA = (_Float16*)smem + QW_STAGE_HALVES + (size_t)wave * QX_PER_WAVE;
    _Float16* bufB = bufA + 16 * AEV_LEN;
    const int atom0 = (blockIdx.x * 8 + wave) * 16;

    // kick off the layer-1 weight DMA early; it overlaps the activation load
    if (wave == 0) tdm_load_blob(w0s, wLds, 12 * 10 * 512 / 2);  // 30720 dwords

    // load 16 x 384 aev tile, f32 -> f16, vectorized float4 (per-wave buffer)
    const float4* src = (const float4*)(aev + (size_t)atom0 * AEV_LEN);
    for (int i = lane; i < 16 * AEV_LEN / 4; i += 32) {
        float4 t = src[i];
        _Float16* d = bufA + i * 4;
        d[0] = (_Float16)t.x; d[1] = (_Float16)t.y;
        d[2] = (_Float16)t.z; d[3] = (_Float16)t.w;
    }
    asm volatile("s_wait_dscnt 0" ::: "memory");
    if (wave == 0) __builtin_amdgcn_s_wait_tensorcnt(0);
    __syncthreads();

    wmma_layer_t<12, 10, true>(bufA, wLds, b0, bufB, lane);    // 384 -> 160
    __syncthreads();
    stage_weights_async(w1s, wLds, 5 * 8 * 512, tid, 256);
    __syncthreads();
    wmma_layer_t<5, 8, true>(bufB, wLds, b1, bufA, lane);      // 160 -> 128
    __syncthreads();
    stage_weights_async(w2s, wLds, 4 * 6 * 512, tid, 256);
    __syncthreads();
    wmma_layer_t<4, 6, true>(bufA, wLds, b2, bufB, lane);      // 128 -> 96

    if (lane < 16) {                       // final 96 -> 1, one atom per lane
        const int atom = atom0 + lane;
        float s = b3[0];
        const _Float16* h = bufB + lane * 96;
        for (int k = 0; k < 96; ++k) s += (float)h[k] * w3[k];
        qAll[atom] = (species[atom] != -1) ? s : 0.f;
    }
}

// ---------------------------------------------------------------------------
// Kernel 2: charge redistribution + species passthrough. One block per molecule.
// ---------------------------------------------------------------------------
__global__ __launch_bounds__(64) void charges_kernel(
    const int* __restrict__ species, const float* __restrict__ qAll,
    const float* __restrict__ netCharge,
    int* __restrict__ outSpecies, float* __restrict__ outPred) {
    __shared__ float s1[64], s2[64];
    const int mol = blockIdx.x, t = threadIdx.x;
    const int atom = mol * NATOM + t;
    const float q = qAll[atom];
    s1[t] = q;
    s2[t] = q * q;
    __syncthreads();
    for (int off = 32; off > 0; off >>= 1) {
        if (t < off) { s1[t] += s1[t + off]; s2[t] += s2[t + off]; }
        __syncthreads();
    }
    const float sumq  = s1[0];
    const float denom = fmaxf(s2[0], 1e-8f);
    const float pred  = q + (netCharge[mol] - sumq) * (q * q) / denom;
    outPred[atom]    = pred;
    outSpecies[atom] = species[atom];
}

// ---------------------------------------------------------------------------
// Kernel 3: ESP + Coulomb energy. One block (64 threads) per molecule.
// Stores energy (initializes the energy output each call).
// ---------------------------------------------------------------------------
__global__ __launch_bounds__(64) void esp_kernel(
    const int* __restrict__ species, const float* __restrict__ coords,
    const float* __restrict__ pred,
    float* __restrict__ espAll, float* __restrict__ energyOut) {
    __shared__ float cx[64], cy[64], cz[64], qv[64], red[64];
    const int mol = blockIdx.x, j = threadIdx.x;
    const int atom = mol * NATOM + j;
    const float x = coords[atom * 3 + 0];
    const float y = coords[atom * 3 + 1];
    const float z = coords[atom * 3 + 2];
    cx[j] = x; cy[j] = y; cz[j] = z;
    qv[j] = pred[atom];                    // zero for padding atoms
    const int spj = species[atom];
    __syncthreads();
    float e = 0.f;
    for (int i = 0; i < NATOM; ++i) {
        if (i == j) continue;
        const float dx = cx[i] - x, dy = cy[i] - y, dz = cz[i] - z;
        const float r2 = dx * dx + dy * dy + dz * dz + 1e-16f;
        e += qv[i] * __frsqrt_rn(r2);
    }
    e *= A0_CONST;                         // q_i / (sqrt(r2)/A0)
    if (spj == -1) e = 0.f;                // pair_mask on receiving atom
    espAll[atom] = e;
    red[j] = qv[j] * e;
    __syncthreads();
    for (int off = 32; off > 0; off >>= 1) {
        if (j < off) red[j] += red[j + off];
        __syncthreads();
    }
    if (j == 0) energyOut[mol] = 0.5f * red[0];   // STORE: initializes output
}

// ---------------------------------------------------------------------------
// Kernel 4: species-routed ANI nets (416pad -> 160 -> 128 -> 96 -> 1).
// Weights staged per WG in LDS (TDM for layer 1, async for layers 2/3);
// per-net skip is workgroup-uniform.
// LDS: 130 KB weight stage + 8 x 22 KB activation buffers = 306 KB.
// ---------------------------------------------------------------------------
struct AniNet {
    const _Float16* w0s; const _Float16* w1s; const _Float16* w2s;
    const float* b0; const float* b1; const float* b2;
    const float* w3; const float* b3;
};
struct AniParams { AniNet n[4]; };

#define AW_STAGE_HALVES   66560                              // max layer: 13*10*512
#define AX_PER_WAVE       (16 * (FEAT_PAD + 160 + 128))      // 11264 halves

__global__ __launch_bounds__(256) void ani_mlp_kernel(
    const float* __restrict__ aev, const int* __restrict__ species,
    const float* __restrict__ predQ, const float* __restrict__ espAll,
    AniParams P, float* __restrict__ energyOut) {
    extern __shared__ char smem[];
    __shared__ int presentMask;
    _Float16* wLds = (_Float16*)smem;
    const int tid  = threadIdx.x;
    const int wave = tid >> 5;
    const int lane = tid & 31;
    _Float16* bufA = (_Float16*)smem + AW_STAGE_HALVES + (size_t)wave * AX_PER_WAVE;
    _Float16* bufB = bufA + 16 * FEAT_PAD;
    _Float16* bufC = bufB + 16 * 160;
    const int atom0 = (blockIdx.x * 8 + wave) * 16;

    if (tid == 0) presentMask = 0;
    __syncthreads();

    // features: [aev(384) | q | esp | 0-pad to 416]
    for (int i = lane; i < 16 * (AEV_LEN / 4); i += 32) {
        const int m = i / (AEV_LEN / 4);
        const int c = i % (AEV_LEN / 4);
        float4 t = *(const float4*)(aev + (size_t)(atom0 + m) * AEV_LEN + c * 4);
        _Float16* d = bufA + m * FEAT_PAD + c * 4;
        d[0] = (_Float16)t.x; d[1] = (_Float16)t.y;
        d[2] = (_Float16)t.z; d[3] = (_Float16)t.w;
    }
    int sp = -2;
    if (lane < 16) {
        const int atom = atom0 + lane;
        sp = species[atom];
        _Float16* row = bufA + lane * FEAT_PAD;
        row[AEV_LEN + 0] = (_Float16)predQ[atom];
        row[AEV_LEN + 1] = (_Float16)espAll[atom];
        for (int c = AEV_LEN + 2; c < FEAT_PAD; ++c) row[c] = (_Float16)0.f;
        if (sp >= 0) atomicOr(&presentMask, 1 << sp);
    }
    asm volatile("s_wait_dscnt 0" ::: "memory");
    __syncthreads();                                   // presentMask + features ready
    const int pm = presentMask;

    float yOut = 0.f;
    for (int net = 0; net < 4; ++net) {
        if (!(pm & (1 << net))) continue;              // WG-uniform: species absent
        const AniNet& N = P.n[net];
        if (wave == 0) {
            tdm_load_blob(N.w0s, wLds, 13 * 10 * 512 / 2);   // 33280 dwords
            __builtin_amdgcn_s_wait_tensorcnt(0);
        }
        __syncthreads();
        wmma_layer_t<13, 10, true>(bufA, wLds, N.b0, bufB, lane);  // 416 -> 160
        __syncthreads();
        stage_weights_async(N.w1s, wLds, 5 * 8 * 512, tid, 256);
        __syncthreads();
        wmma_layer_t<5, 8, true>(bufB, wLds, N.b1, bufC, lane);    // 160 -> 128
        __syncthreads();
        stage_weights_async(N.w2s, wLds, 4 * 6 * 512, tid, 256);
        __syncthreads();
        wmma_layer_t<4, 6, true>(bufC, wLds, N.b2, bufB, lane);    // 128 -> 96
        if (lane < 16) {
            float s = N.b3[0];
            const _Float16* h = bufB + lane * 96;
            for (int k = 0; k < 96; ++k) s += (float)h[k] * N.w3[k];
            if (sp == net) yOut = s;
        }
        __syncthreads();                               // B reads done before restage
    }
    if (lane < 16) {
        const int atom = atom0 + lane;
        atomicAdd(&energyOut[atom >> 6], yOut);        // mol energy on top of coulomb
    }
}

// ---------------------------------------------------------------------------
// Host launcher
// ---------------------------------------------------------------------------
extern "C" void kernel_launch(void* const* d_in, const int* in_sizes, int n_in,
                              void* d_out, int out_size, void* d_ws, size_t ws_size,
                              hipStream_t stream) {
    (void)in_sizes; (void)n_in; (void)out_size; (void)ws_size;

    const int*   species   = (const int*)d_in[0];
    const float* coords    = (const float*)d_in[1];
    const float* aev       = (const float*)d_in[2];
    const float* netCharge = (const float*)d_in[3];

    // ---- workspace layout ----
    // [0, 1MB): f16 swizzled weights; [1MB, ...): float scratch
    _Float16* wsH = (_Float16*)d_ws;
    _Float16* qw0 = wsH;                    // 12*10*512 = 61440
    _Float16* qw1 = qw0 + 61440;            //  5* 8*512 = 20480
    _Float16* qw2 = qw1 + 20480;            //  4* 6*512 = 12288
    _Float16* aw0[4]; _Float16* aw1[4]; _Float16* aw2[4];
    size_t aniOff = 94208;
    for (int i = 0; i < 4; ++i) {
        aw0[i] = wsH + aniOff;              // 13*10*512 = 66560
        aw1[i] = aw0[i] + 66560;            //  5* 8*512 = 20480
        aw2[i] = aw1[i] + 20480;            //  4* 6*512 = 12288
        aniOff += 99328;
    }
    float* wsF    = (float*)((char*)d_ws + (1u << 20));
    float* qAll   = wsF;                    // NATOMS
    float* espAll = wsF + NATOMS;           // NATOMS

    // ---- output layout: species(int,131072) | energy(2048) | pred(131072) ----
    int*   outSpecies = (int*)d_out;
    float* outEnergy  = (float*)d_out + NATOMS;
    float* outPred    = outEnergy + NMOL;

    // ---- 1) weight swizzle (f32 -> f16 fragment-order) ----
    auto launchSwz = [&](const void* W, _Float16* dst, int din, int dout, int ks) {
        const int total = ks * (dout / 16) * 512;
        swizzle_weights_kernel<<<(total + 255) / 256, 256, 0, stream>>>(
            (const float*)W, dst, din, dout, ks);
    };
    launchSwz(d_in[4], qw0, 384, 160, 12);
    launchSwz(d_in[6], qw1, 160, 128, 5);
    launchSwz(d_in[8], qw2, 128, 96, 4);
    for (int i = 0; i < 4; ++i) {
        const int b = 12 + 8 * i;
        launchSwz(d_in[b + 0], aw0[i], 386, 160, 13);   // K-padded to 416
        launchSwz(d_in[b + 2], aw1[i], 160, 128, 5);
        launchSwz(d_in[b + 4], aw2[i], 128, 96, 4);
    }

    // ---- 2) q-net MLP ----
    const size_t shQ = (size_t)(QW_STAGE_HALVES + 8 * QX_PER_WAVE) * sizeof(_Float16); // 256 KB
    (void)hipFuncSetAttribute((const void*)q_mlp_kernel,
                              hipFuncAttributeMaxDynamicSharedMemorySize, (int)shQ);
    q_mlp_kernel<<<NATOMS / 128, 256, shQ, stream>>>(
        aev, species,
        qw0, (const float*)d_in[5], qw1, (const float*)d_in[7],
        qw2, (const float*)d_in[9], (const float*)d_in[10], (const float*)d_in[11],
        qAll);

    // ---- 3) charge redistribution ----
    charges_kernel<<<NMOL, 64, 0, stream>>>(species, qAll, netCharge, outSpecies, outPred);

    // ---- 4) ESP + coulomb energy (stores energy) ----
    esp_kernel<<<NMOL, 64, 0, stream>>>(species, coords, outPred, espAll, outEnergy);

    // ---- 5) species-routed ANI nets (atomic-adds atomic energies) ----
    AniParams P;
    for (int i = 0; i < 4; ++i) {
        const int b = 12 + 8 * i;
        P.n[i].w0s = aw0[i]; P.n[i].w1s = aw1[i]; P.n[i].w2s = aw2[i];
        P.n[i].b0 = (const float*)d_in[b + 1];
        P.n[i].b1 = (const float*)d_in[b + 3];
        P.n[i].b2 = (const float*)d_in[b + 5];
        P.n[i].w3 = (const float*)d_in[b + 6];
        P.n[i].b3 = (const float*)d_in[b + 7];
    }
    const size_t shA = (size_t)(AW_STAGE_HALVES + 8 * AX_PER_WAVE) * sizeof(_Float16); // 306 KB
    (void)hipFuncSetAttribute((const void*)ani_mlp_kernel,
                              hipFuncAttributeMaxDynamicSharedMemorySize, (int)shA);
    ani_mlp_kernel<<<NATOMS / 128, 256, shA, stream>>>(
        aev, species, outPred, espAll, P, outEnergy);
}
